// DigitConvolutionalModel_41326175322903
// MI455X (gfx1250) — matrix operand order repaired and checked by reading, not compiled, not run on hardware
//
#include <hip/hip_runtime.h>

typedef __attribute__((ext_vector_type(16))) _Float16 v16h;
typedef __attribute__((ext_vector_type(8)))  _Float16 v8h;
typedef __attribute__((ext_vector_type(8)))  float    v8f;

#define TILE_B 32
#define NWAVES 8
#define K1 704   // 676 padded to multiple of 32
#define N1 128
#define N2 512
#define N3 256
#define N4P 16   // 10 padded to 16
#define K4 256

// ---------------------------------------------------------------------------
// Weight conversion: fp32 -> f16, row-major [n][k], K padded with zeros.
// Layout in d_ws (halves): w1h[128*704] | w2h[512*128] | w3h[256*512] | w4h[16*256]
// ---------------------------------------------------------------------------
__global__ void convert_weights_kernel(const float* __restrict__ w1,
                                       const float* __restrict__ w2,
                                       const float* __restrict__ w3,
                                       const float* __restrict__ w4,
                                       _Float16* __restrict__ w1h,
                                       _Float16* __restrict__ w2h,
                                       _Float16* __restrict__ w3h,
                                       _Float16* __restrict__ w4h) {
  int idx = blockIdx.x * blockDim.x + threadIdx.x;
  if (idx < 128 * K1) {
    int n = idx / K1, k = idx % K1;
    w1h[idx] = (_Float16)(k < 676 ? w1[n * 676 + k] : 0.0f);
  }
  if (idx < 512 * 128) w2h[idx] = (_Float16)w2[idx];
  if (idx < 256 * 512) w3h[idx] = (_Float16)w3[idx];
  if (idx < N4P * K4) {
    int n = idx / K4, k = idx % K4;
    w4h[idx] = (_Float16)(n < 10 ? w4[n * K4 + k] : 0.0f);
  }
}

// ---------------------------------------------------------------------------
// One 16x16 output tile, accumulated over K with v_wmma_f32_16x16x32_f16.
// A (16xK f16) in LDS, row stride = K. A layout used here (contiguous k, not AMD's table order):
//   lanes 0-15 : M=lane,    elems 0-15 = K kb+0..15
//   lanes16-31 : M=lane-16, elems 0-15 = K kb+16..31
// B (Kx16, == weights row-major [n][k]) in global (L2-hot). B layout (same contiguous k as A):
//   lanes 0-15 : N=lane,    elems 0-15 = K kb+0..15
//   lanes16-31 : N=lane-16, elems 0-15 = K kb+16..31
// ---------------------------------------------------------------------------
template <int K>
__device__ __forceinline__ v8f wmma_tile_acc(const _Float16* __restrict__ act,
                                             const _Float16* __restrict__ wgt,
                                             int m0, int n0, int lane) {
  const int l15  = lane & 15;
  const int half = lane >> 4;                 // 0 or 1
  const _Float16* arow = act + (m0 + l15) * K + half * 16;
  const _Float16* brow = wgt + (n0 + l15) * K + half * 16;
  v8f acc = {};
#pragma unroll
  for (int kb = 0; kb < K; kb += 32) {
    v8h alo = *(const v8h*)(arow + kb);
    v8h ahi = *(const v8h*)(arow + kb + 8);
    v16h a = __builtin_shufflevector(alo, ahi,
                                     0, 1, 2, 3, 4, 5, 6, 7,
                                     8, 9, 10, 11, 12, 13, 14, 15);
    v16h b = *(const v16h*)(brow + kb);
    acc = __builtin_amdgcn_wmma_f32_16x16x32_f16(
        /*neg_a=*/false, a, /*neg_b=*/false, b,
        /*c_mod=*/(short)0, acc, /*reuse_a=*/false, /*reuse_b=*/false);
  }
  return acc;
}

// C/D layout: VGPR g: lanes 0-15 -> (M=m0+g, N=n0+lane); lanes 16-31 -> M=m0+g+8.
template <int N, bool RELU>
__device__ __forceinline__ void store_tile_lds(v8f acc,
                                               const float* __restrict__ bias,
                                               _Float16* __restrict__ out,
                                               int m0, int n0, int lane) {
  const int n = n0 + (lane & 15);
  const float bv = bias[n];
  const int mbase = m0 + (lane >> 4) * 8;
#pragma unroll
  for (int g = 0; g < 8; ++g) {
    float v = acc[g] + bv;
    if (RELU && v < 0.0f) v = 0.0f;
    out[(mbase + g) * N + n] = (_Float16)v;
  }
}

template <int K, int N, bool RELU>
__device__ __forceinline__ void dense_layer(const _Float16* __restrict__ act,
                                            const _Float16* __restrict__ wgt,
                                            const float* __restrict__ bias,
                                            _Float16* __restrict__ out,
                                            int wid, int lane) {
  constexpr int MT = TILE_B / 16;
  constexpr int NT = N / 16;
  for (int t = wid; t < MT * NT; t += NWAVES) {   // wave-uniform loop
    const int m0 = (t % MT) * 16;
    const int n0 = (t / MT) * 16;
    v8f acc = wmma_tile_acc<K>(act, wgt, m0, n0, lane);
    store_tile_lds<N, RELU>(acc, bias, out, m0, n0, lane);
  }
}

// ---------------------------------------------------------------------------
// Fused conv + 4-layer MLP. One block = TILE_B batch rows, 8 waves.
// ---------------------------------------------------------------------------
__global__ void __launch_bounds__(256)
fused_mlp_kernel(const float* __restrict__ x,
                 const float* __restrict__ wconv,
                 const _Float16* __restrict__ w1h, const float* __restrict__ b1,
                 const _Float16* __restrict__ w2h, const float* __restrict__ b2,
                 const _Float16* __restrict__ w3h, const float* __restrict__ b3,
                 const _Float16* __restrict__ w4h, const float* __restrict__ b4,
                 float* __restrict__ out) {
  __shared__ _Float16 bufX[TILE_B * K1];   // conv acts (K=704) -> later h2 (32x512)
  __shared__ _Float16 bufY[TILE_B * 256];  // h1 (32x128) -> later h3 (32x256)

  const int tid  = threadIdx.x;
  const int lane = tid & 31;
  const int wid  = tid >> 5;
  const size_t row0 = (size_t)blockIdx.x * TILE_B;

  // ---- Phase 0: 3x3 valid conv, 28x28 -> 26x26, f16 into bufX (padded) ----
  float wc[9];
#pragma unroll
  for (int i = 0; i < 9; ++i) wc[i] = wconv[i];

  const float* xblk = x + row0 * 784;
  for (int idx = tid; idx < TILE_B * K1; idx += 256) {
    const int b = idx / K1;
    const int p = idx % K1;
    float v = 0.0f;
    if (p < 676) {
      const int r = p / 26, c = p % 26;
      const float* xr = xblk + b * 784 + r * 28 + c;
      v = xr[0]  * wc[0] + xr[1]  * wc[1] + xr[2]  * wc[2] +
          xr[28] * wc[3] + xr[29] * wc[4] + xr[30] * wc[5] +
          xr[56] * wc[6] + xr[57] * wc[7] + xr[58] * wc[8];
    }
    bufX[idx] = (_Float16)v;
  }
  __syncthreads();

  // ---- Layer 1: [32,704] x [704,128] -> relu -> bufY ----
  dense_layer<K1, N1, true>(bufX, w1h, b1, bufY, wid, lane);
  __syncthreads();

  // ---- Layer 2: [32,128] x [128,512] -> relu -> bufX (acts dead) ----
  dense_layer<N1, N2, true>(bufY, w2h, b2, bufX, wid, lane);
  __syncthreads();

  // ---- Layer 3: [32,512] x [512,256] -> relu -> bufY (h1 dead) ----
  dense_layer<N2, N3, true>(bufX, w3h, b3, bufY, wid, lane);
  __syncthreads();

  // ---- Layer 4: [32,256] x [256,16pad] -> fp32 + bias -> global [B,10] ----
  for (int t = wid; t < TILE_B / 16; t += NWAVES) {   // wave-uniform
    const int m0 = t * 16;
    v8f acc = wmma_tile_acc<K4>(bufY, w4h, m0, 0, lane);
    const int n = lane & 15;
    const int mbase = m0 + (lane >> 4) * 8;
    if (n < 10) {                                     // divergence only after WMMA
      const float bv = b4[n];
#pragma unroll
      for (int g = 0; g < 8; ++g) {
        out[(row0 + mbase + g) * 10 + n] = acc[g] + bv;
      }
    }
  }
}

// ---------------------------------------------------------------------------
extern "C" void kernel_launch(void* const* d_in, const int* in_sizes, int n_in,
                              void* d_out, int out_size, void* d_ws, size_t ws_size,
                              hipStream_t stream) {
  const float* x     = (const float*)d_in[0];
  const float* wconv = (const float*)d_in[1];
  const float* w1    = (const float*)d_in[2];
  const float* b1    = (const float*)d_in[3];
  const float* w2    = (const float*)d_in[4];
  const float* b2    = (const float*)d_in[5];
  const float* w3    = (const float*)d_in[6];
  const float* b3    = (const float*)d_in[7];
  const float* w4    = (const float*)d_in[8];
  const float* b4    = (const float*)d_in[9];
  float* out = (float*)d_out;

  _Float16* w1h = (_Float16*)d_ws;
  _Float16* w2h = w1h + 128 * K1;
  _Float16* w3h = w2h + 512 * 128;
  _Float16* w4h = w3h + 256 * 512;

  // Convert weights (covers the largest array: 256*512 = 131072 elements).
  convert_weights_kernel<<<(131072 + 255) / 256, 256, 0, stream>>>(
      w1, w2, w3, w4, w1h, w2h, w3h, w4h);

  const int B = in_sizes[0] / 784;
  fused_mlp_kernel<<<B / TILE_B, 256, 0, stream>>>(
      x, wconv, w1h, b1, w2h, b2, w3h, b3, w4h, b4, out);
}
